// SeparateWeightTensorProduct_71554155151877
// MI455X (gfx1250) — matrix operand order repaired and checked by reading, not compiled, not run on hardware
//
#include <hip/hip_runtime.h>

typedef __attribute__((ext_vector_type(16))) __bf16 v16bf;
typedef __attribute__((ext_vector_type(8)))  __bf16 v8bf;
typedef __attribute__((ext_vector_type(8)))  float  v8f;

#define BATCH 32768
#define DIM1  1152
#define MT    32      // batch rows per workgroup
#define NTH   512     // 16 waves (wave32)

// ---------------- instruction metadata (11 allowed uvw paths) ----------------
// idx:        0  1  2  3  4  5  6  7  8  9 10
constexpr int C_L1[11] = {0, 0, 0, 1, 1, 1, 1, 2, 2, 2, 2};
constexpr int C_L2[11] = {0, 1, 2, 0, 1, 1, 2, 0, 1, 2, 2};
constexpr int C_LO[11] = {0, 1, 2, 1, 0, 2, 1, 2, 1, 0, 2};
// column base of each instruction inside its i1-group of T columns
constexpr int C_CB[11] = {0, 1, 4, 0, 3, 4, 9, 0, 5, 8, 9};
constexpr int C_NC[3]    = {9, 12, 14};   // T columns per i1 group
constexpr int C_GBASE[3] = {0, 9, 45};    // flat G row base per i1 group (total 115)
constexpr int C_OFF[3]   = {0, 128, 512}; // OFF1 == OFFO block offsets
constexpr int C_OFF2[3]  = {0, 1, 4};     // x2 block offsets
constexpr int C_OCB[3]   = {0, 1, 4};     // accumulator base per io

// =====================================================================
// build_gb: exact real-basis Wigner-3j * path-weight -> GB[115][9]
// G[b] = GB @ x2[b].  Computed on device (graph-capture safe, deterministic).
// =====================================================================
struct Cpx { float re, im; };
__device__ inline Cpx cmul(Cpx a, Cpx b){ return Cpx{a.re*b.re - a.im*b.im, a.re*b.im + a.im*b.re}; }
__device__ inline float ffact(int n){ float r = 1.f; for (int i = 2; i <= n; ++i) r *= (float)i; return r; }

__device__ float cgc(int j1,int m1,int j2,int m2,int j3,int m3){
  int vmin = max(max(-j1 + j2 + m3, -j1 + m1), 0);
  int vmax = min(min(j2 + j3 + m1, j3 - j1 + j2), j3 + m3);
  float C = sqrtf((float)(2*j3+1)*ffact(j3+j1-j2)*ffact(j3-j1+j2)*ffact(j1+j2-j3)
                  *ffact(j3+m3)*ffact(j3-m3)
                  /(ffact(j1+j2+j3+1)*ffact(j1-m1)*ffact(j1+m1)*ffact(j2-m2)*ffact(j2+m2)));
  float S = 0.f;
  for (int v = vmin; v <= vmax; ++v){
    float sgn = ((v + j2 + m2) & 1) ? -1.f : 1.f;
    S += sgn*ffact(j2+j3+m1-v)*ffact(j1-m1+v)
        /(ffact(v)*ffact(j3-j1+j2-v)*ffact(j3+m3-v)*ffact(v+j1-j2-m3));
  }
  return C*S;
}

__device__ void qfill(int l, Cpx* q){   // real->complex SH change of basis
  int n = 2*l + 1;
  for (int a = 0; a < n*n; ++a) q[a] = Cpx{0.f, 0.f};
  const float is2 = 0.70710678118654752f;
  for (int m = -l; m < 0; ++m){
    q[(l+m)*n + (l-m)] = Cpx{is2, 0.f};
    q[(l+m)*n + (l+m)] = Cpx{0.f, -is2};
  }
  q[l*n + l] = Cpx{1.f, 0.f};
  for (int m = 1; m <= l; ++m){
    float s = (m & 1) ? -1.f : 1.f;
    q[(l+m)*n + (l+m)] = Cpx{s*is2, 0.f};
    q[(l+m)*n + (l-m)] = Cpx{0.f, s*is2};
  }
  Cpx ph{1.f, 0.f};
  for (int t = 0; t < l; ++t) ph = cmul(ph, Cpx{0.f, -1.f});
  for (int a = 0; a < n*n; ++a) q[a] = cmul(ph, q[a]);
}

__global__ void build_gb(float* __restrict__ GB)
{
  __shared__ float cg[125];
  __shared__ Cpx q1s[25], q2s[25], q3s[25];
  __shared__ float vals[125];
  __shared__ float nrm2;

  const int idx = blockIdx.x;
  const int t   = threadIdx.x;
  const int l1 = C_L1[idx], l2 = C_L2[idx], l3 = C_LO[idx];
  const int d1 = 2*l1+1, d2 = 2*l2+1, d3 = 2*l3+1;
  const int ne = d1*d2*d3;

  if (t == 0){ qfill(l1, q1s); qfill(l2, q2s); qfill(l3, q3s); nrm2 = 0.f; }
  for (int e = t; e < ne; e += blockDim.x){
    int i = e/(d2*d3), r = e%(d2*d3), k = r/d3, n = r%d3;
    int m1 = i - l1, m2 = k - l2, m3 = n - l3;
    cg[e] = (m1 + m2 == m3) ? cgc(l1, m1, l2, m2, l3, m3) : 0.f;
  }
  __syncthreads();
  // result[j,L,m] = sum_{i,k,n} q1[i,j] q2[k,L] conj(q3[n,m]) cg[i,k,n]; take real part
  for (int e = t; e < ne; e += blockDim.x){
    int j = e/(d2*d3), r = e%(d2*d3), L = r/d3, m = r%d3;
    Cpx s{0.f, 0.f};
    for (int i = 0; i < d1; ++i)
      for (int k = 0; k < d2; ++k){
        Cpx qa = cmul(q1s[i*d1 + j], q2s[k*d2 + L]);
        for (int n = 0; n < d3; ++n){
          float c = cg[(i*d2 + k)*d3 + n];
          Cpx q3c{q3s[n*d3 + m].re, -q3s[n*d3 + m].im};
          Cpx p = cmul(qa, q3c);
          s.re += p.re*c;
        }
      }
    vals[e] = s.re;
    atomicAdd(&nrm2, s.re*s.re);
  }
  __syncthreads();
  // path weights: io=0: sqrt(1/384), io=1: sqrt(3/512), io=2: sqrt(5/512)
  const float PW[3] = {0.05103103630798288f, 0.07654655446197431f, 0.09882117688026186f};
  const float scale = PW[l3] * rsqrtf(nrm2);
  for (int e = t; e < d1*d3; e += blockDim.x){
    int i = e/d3, k = e%d3;
    int g = C_GBASE[l1] + i*C_NC[l1] + C_CB[idx] + k;
    #pragma unroll
    for (int j = 0; j < 9; ++j) GB[g*9 + j] = 0.f;
    for (int j = 0; j < d2; ++j) GB[g*9 + C_OFF2[l2] + j] = vals[(i*d2 + j)*d3 + k]*scale;
  }
}

// =====================================================================
// build_wt: Weff[idx][u][w] = w1[idx][u][w]*w2[idx][0][w], stored transposed
// (w-major, u contiguous) in bf16 so B fragments are contiguous 32B loads.
// =====================================================================
__global__ void build_wt(const float* __restrict__ w1, const float* __restrict__ w2,
                         __bf16* __restrict__ Wt)
{
  int t = blockIdx.x*256 + threadIdx.x;
  if (t >= 11*128*128) return;
  int idx = t >> 14;
  int rem = t & 16383;
  int w = rem >> 7;
  int u = rem & 127;
  float v = w1[(size_t)idx*16384 + u*128 + w] * w2[idx*128 + w];
  Wt[t] = (__bf16)v;   // t == idx*16384 + w*128 + u
}

// =====================================================================
// tp_main: fused kernel.
//   per WG: 32 batch rows x full 128 w.  16 waves = 2(M) x 8(N) of 16x16 tiles.
//   per path: T tile (32 x d_o x 128) -> LDS (bf16), then
//   acc[oc] += T_k (16x128) @ Weff (128x16) via 4 chained bf16 WMMAs.
// =====================================================================
__global__ __launch_bounds__(NTH)
void tp_main(const float* __restrict__ x1, const float* __restrict__ x2,
             const __bf16* __restrict__ Wt, const float* __restrict__ GB,
             float* __restrict__ out)
{
  __shared__ float x2s[MT*9];
  __shared__ float GBs[115*9];
  __shared__ float Gs[MT*120];
  __shared__ __align__(16) __bf16 Tlds[5*MT*128];   // 40 KB

  const int tid   = threadIdx.x;
  const int bbase = blockIdx.x * MT;

  for (int e = tid; e < MT*9;  e += NTH) x2s[e] = x2[bbase*9 + e];
  for (int e = tid; e < 115*9; e += NTH) GBs[e] = GB[e];
  __syncthreads();
  for (int e = tid; e < MT*115; e += NTH){
    int bl = e/115, g = e - bl*115;
    float s = 0.f;
    #pragma unroll
    for (int j = 0; j < 9; ++j) s += GBs[g*9 + j]*x2s[bl*9 + j];
    Gs[bl*120 + g] = s;
  }
  __syncthreads();

  const int lane = tid & 31;
  const int wv   = tid >> 5;       // 0..15
  const int mp   = wv & 1;         // M tile position (2 x 16 rows)
  const int np   = wv >> 1;        // N tile position (8 x 16 cols)
  const int blr  = tid >> 4;       // T-compute batch row (0..31)
  const int u0   = (tid & 15)*8;   // T-compute u base (8 contiguous u per thread)

  v8f acc[9];
  #pragma unroll
  for (int o = 0; o < 9; ++o){
    #pragma unroll
    for (int r = 0; r < 8; ++r) acc[o][r] = 0.f;
  }

  float xr[8][5];

  #pragma unroll
  for (int idx = 0; idx < 11; ++idx){
    const int l1  = C_L1[idx];
    const int d1  = 2*l1 + 1;
    const int loO = C_LO[idx];
    const int d_o = 2*loO + 1;
    const int nc  = C_NC[l1];
    const int gb0 = C_GBASE[l1] + C_CB[idx];

    // register-cache this i1 block of x1 once per group (read exactly once)
    if (idx == 0 || idx == 3 || idx == 7){
      const float* xp = x1 + (size_t)(bbase + blr)*DIM1 + C_OFF[l1] + u0*d1;
      #pragma unroll
      for (int p = 0; p < 8; ++p){
        #pragma unroll
        for (int i = 0; i < d1; ++i) xr[p][i] = xp[p*d1 + i];
      }
    }

    // G values for this (batch row, instruction): broadcast LDS reads
    float garr[5][5];
    #pragma unroll
    for (int i = 0; i < d1; ++i){
      #pragma unroll
      for (int k = 0; k < d_o; ++k) garr[i][k] = Gs[blr*120 + gb0 + i*nc + k];
    }

    // T tile -> LDS as bf16, packed 8-wide stores
    #pragma unroll
    for (int k = 0; k < d_o; ++k){
      v8bf pk;
      #pragma unroll
      for (int p = 0; p < 8; ++p){
        float s = 0.f;
        #pragma unroll
        for (int i = 0; i < d1; ++i) s += xr[p][i]*garr[i][k];
        pk[p] = (__bf16)s;
      }
      *(v8bf*)(&Tlds[(k*MT + blr)*128 + u0]) = pk;
    }
    __syncthreads();

    // B fragments (32x16, K=u, N=w) from transposed bf16 weights (L2-hot).
    // Layout: lanes 0-15 hold K 0..15, lanes 16-31 hold K 16..31; N = lane&15.
    v16bf bfrag[4];
    {
      const int w  = np*16 + (lane & 15);
      const int kb = (lane & 16) ? 16 : 0;
      const __bf16* wp = Wt + (size_t)idx*16384 + w*128 + kb;
      #pragma unroll
      for (int ks = 0; ks < 4; ++ks)
        bfrag[ks] = *(const v16bf*)(wp + ks*32);
    }

    // WMMA: acc[ocb+k] += T_k(16x128) @ Weff(128x16), K split into 4x32
    {
      const int row = mp*16 + (lane & 15);
      const int kb  = (lane & 16) ? 16 : 0;   // A layout: lanes<16 K 0-15, >=16 K 16-31
      #pragma unroll
      for (int k = 0; k < d_o; ++k){
        const __bf16* tp = &Tlds[(k*MT + row)*128];
        #pragma unroll
        for (int ks = 0; ks < 4; ++ks){
          v8bf lo8 = *(const v8bf*)(tp + ks*32 + kb);
          v8bf hi8 = *(const v8bf*)(tp + ks*32 + kb + 8);
          v16bf af;
          #pragma unroll
          for (int e = 0; e < 8; ++e){ af[e] = lo8[e]; af[e + 8] = hi8[e]; }
          acc[C_OCB[loO] + k] = __builtin_amdgcn_wmma_f32_16x16x32_bf16(
              false, af, false, bfrag[ks], (short)0, acc[C_OCB[loO] + k], false, false);
        }
      }
    }
    __syncthreads();
  }

  // store: C/D layout -> lane&15 = N(w), VGPR r = M row r (+8 for lanes>=16)
  const int w   = np*16 + (lane & 15);
  const int mof = mp*16 + ((lane & 16) ? 8 : 0);
  #pragma unroll
  for (int io = 0; io < 3; ++io){
    const int d_o = 2*io + 1;
    #pragma unroll
    for (int k = 0; k < d_o; ++k){
      const v8f a = acc[C_OCB[io] + k];
      #pragma unroll
      for (int r = 0; r < 8; ++r){
        int b = bbase + mof + r;
        out[(size_t)b*DIM1 + C_OFF[io] + w*d_o + k] = a[r];
      }
    }
  }
}

// =====================================================================
extern "C" void kernel_launch(void* const* d_in, const int* in_sizes, int n_in,
                              void* d_out, int out_size, void* d_ws, size_t ws_size,
                              hipStream_t stream)
{
  (void)in_sizes; (void)n_in; (void)out_size; (void)ws_size;
  const float* x1 = (const float*)d_in[0];   // (32768, 1152)
  const float* x2 = (const float*)d_in[1];   // (32768, 9)
  const float* w1 = (const float*)d_in[2];   // (11, 128, 128)
  const float* w2 = (const float*)d_in[3];   // (11, 1, 128)
  float* out = (float*)d_out;                // (32768, 1152)

  float*  dGB = (float*)d_ws;                        // 115*9 floats
  __bf16* dWt = (__bf16*)((char*)d_ws + 16384);      // 11*128*128 bf16 (360 KB)

  build_gb<<<11, 128, 0, stream>>>(dGB);
  build_wt<<<704, 256, 0, stream>>>(w1, w2, dWt);
  tp_main<<<BATCH/MT, NTH, 0, stream>>>(x1, x2, dWt, dGB, out);
}
